// Block_58016418234825
// MI455X (gfx1250) — matrix operand order repaired and checked by reading, not compiled, not run on hardware
//
#include <hip/hip_runtime.h>
#include <hip/hip_bf16.h>

// ---------------------------------------------------------------------------
// Types for CDNA5 WMMA (wave32): v_wmma_f32_16x16x32_bf16
// ---------------------------------------------------------------------------
typedef __bf16 bf16_t;
typedef __attribute__((ext_vector_type(16))) __bf16 v16bf;
typedef __attribute__((ext_vector_type(8)))  float  v8f;

union FragBF {
    v16bf v;
    uint4 q[2];
};

__device__ __forceinline__ v8f wmma_bf16(v16bf a, v16bf b, v8f c) {
    // 8 args: (neg_a, A, neg_b, B, c_mod, C, reuse_a, reuse_b)
    return __builtin_amdgcn_wmma_f32_16x16x32_bf16(false, a, false, b, (short)0, c, false, false);
}

__device__ __forceinline__ v8f v8f_zero() {
    v8f z;
#pragma unroll
    for (int i = 0; i < 8; ++i) z[i] = 0.f;
    return z;
}

__device__ __forceinline__ unsigned short f2bf(float f) {
    unsigned int u = __float_as_uint(f);
    u += 0x7FFFu + ((u >> 16) & 1u);           // round-to-nearest-even
    return (unsigned short)(u >> 16);
}
__device__ __forceinline__ float bf2f(unsigned short h) {
    return __uint_as_float(((unsigned int)h) << 16);
}
__device__ __forceinline__ unsigned int pack2(float a, float b) {
    return (unsigned int)f2bf(a) | ((unsigned int)f2bf(b) << 16);
}
__device__ __forceinline__ unsigned int get16(const uint4& q, int n) {
    const unsigned int* w = (const unsigned int*)&q;
    return (w[n >> 1] >> ((n & 1) * 16)) & 0xffffu;
}

// ---------------------------------------------------------------------------
// Problem constants
// ---------------------------------------------------------------------------
constexpr int S_  = 4096;
constexpr int H_  = 2048;
constexpr int NH_ = 16;
constexpr int HD2 = 128;   // head dim
constexpr int I_  = 5504;

// ---------------------------------------------------------------------------
// fp32 -> bf16 conversion (weights), 4 elems/thread
// ---------------------------------------------------------------------------
__global__ __launch_bounds__(256) void cvt_f32_to_bf16(const float* __restrict__ in,
                                                       unsigned short* __restrict__ out,
                                                       int n4) {
    int i = blockIdx.x * 256 + threadIdx.x;
    if (i >= n4) return;
    float4 v = ((const float4*)in)[i];
    uint2 r;
    r.x = pack2(v.x, v.y);
    r.y = pack2(v.z, v.w);
    ((uint2*)out)[i] = r;
}

// ---------------------------------------------------------------------------
// RMSNorm (H=2048) -> bf16; one row per workgroup (8 wave32 waves)
// ---------------------------------------------------------------------------
__global__ __launch_bounds__(256) void rmsnorm_to_bf16(const float* __restrict__ x,
                                                       const float* __restrict__ g,
                                                       unsigned short* __restrict__ out) {
    __shared__ float red[8];
    const int row = blockIdx.x;
    const int tid = threadIdx.x;
    const float* xr = x + (size_t)row * H_;
    float ss = 0.f;
#pragma unroll
    for (int i = 0; i < 8; ++i) {
        float v = xr[tid + 256 * i];
        ss += v * v;
    }
#pragma unroll
    for (int m = 1; m < 32; m <<= 1) ss += __shfl_xor(ss, m, 32);
    if ((tid & 31) == 0) red[tid >> 5] = ss;
    __syncthreads();
    float tot = 0.f;
#pragma unroll
    for (int i = 0; i < 8; ++i) tot += red[i];
    const float rr = rsqrtf(tot * (1.f / (float)H_) + 1e-6f);
    unsigned short* orow = out + (size_t)row * H_;
#pragma unroll
    for (int i = 0; i < 8; ++i) {
        int c = tid + 256 * i;
        orow[c] = f2bf(xr[c] * rr * g[c]);
    }
}

// ---------------------------------------------------------------------------
// RoPE in-place on q & k (bf16, layout (S, NH*HD)); one thread per (s,head,pair)
// ---------------------------------------------------------------------------
__global__ __launch_bounds__(256) void rope_inplace(unsigned short* __restrict__ q,
                                                    unsigned short* __restrict__ k) {
    int idx = blockIdx.x * 256 + threadIdx.x;   // S*NH*64 total
    int i    = idx & 63;
    int head = (idx >> 6) & 15;
    int s    = idx >> 10;
    // inv_freq = 10000^(-2i/128) = exp2(-(2i/128)*log2(10000))
    float freq = exp2f((-2.f / 128.f) * (float)i * 13.287712379549449f);
    float ang = (float)s * freq;
    float sn, cs;
    __sincosf(ang, &sn, &cs);
    size_t base = (size_t)s * H_ + (size_t)head * HD2 + 2 * i;
    {
        float e = bf2f(q[base]), o = bf2f(q[base + 1]);
        *(unsigned int*)(q + base) = pack2(e * cs - o * sn, e * sn + o * cs);
    }
    {
        float e = bf2f(k[base]), o = bf2f(k[base + 1]);
        *(unsigned int*)(k + base) = pack2(e * cs - o * sn, e * sn + o * cs);
    }
}

// ---------------------------------------------------------------------------
// bf16 WMMA GEMM: C(MxN) = A(MxK) @ B(KxN). 128x128 tile, BK=64,
// register-staged double-buffered LDS pipeline (1 barrier / K-step).
// 8 wave32 waves: wave grid 4(m) x 2(n); each wave -> 32x64, 16 WMMA / K-step.
//  MODE 0: outBF = bf16(acc)
//  MODE 1: outF  = acc + resid            (fp32 out, residual add)
//  MODE 2: outBF = bf16(silu(acc) * aux)  (SwiGLU gate fused with up)
// ---------------------------------------------------------------------------
template <int MODE>
__global__ __launch_bounds__(256)
void gemm_bf16(const unsigned short* __restrict__ A,
               const unsigned short* __restrict__ B,
               int M, int N, int K,
               unsigned short* __restrict__ outBF,
               float* __restrict__ outF,
               const float* __restrict__ resid,
               const unsigned short* __restrict__ aux) {
    (void)M;
    constexpr int LD = 72;                        // u16 stride: 64 + 8 pad (144B)
    __shared__ unsigned short As[2][128 * LD];    // [m][k]
    __shared__ unsigned short Bs[2][128 * LD];    // transposed: [n][k]

    const int tid  = threadIdx.x;
    const int lane = tid & 31;
    const int wave = tid >> 5;
    const int wm = wave & 3;                      // 0..3
    const int wn = wave >> 2;                     // 0..1
    const int m0 = blockIdx.y * 128;
    const int n0 = blockIdx.x * 128;

    v8f acc[2][4];
#pragma unroll
    for (int mt = 0; mt < 2; ++mt)
#pragma unroll
        for (int nt = 0; nt < 4; ++nt) acc[mt][nt] = v8f_zero();

    // loader indices
    const int ar = tid >> 1;                      // A tile row 0..127
    const int ak = (tid & 1) * 32;                // A k-chunk (u16)
    const int bkq = (tid >> 4) * 4;               // B: 4 consecutive k rows
    const int bnq = (tid & 15) * 8;               // B: 8 n cols (lanes coalesced)

    uint4 ra[4], rb[4];
    auto loadT = [&](int k0) {
        const uint4* ga = (const uint4*)(A + (size_t)(m0 + ar) * K + k0 + ak);
#pragma unroll
        for (int i = 0; i < 4; ++i) ra[i] = ga[i];
#pragma unroll
        for (int r = 0; r < 4; ++r)
            rb[r] = *(const uint4*)(B + (size_t)(k0 + bkq + r) * N + n0 + bnq);
    };
    auto storeT = [&](int buf) {
        unsigned short* pa = &As[buf][ar * LD + ak];
#pragma unroll
        for (int i = 0; i < 4; ++i) *(uint4*)(pa + 8 * i) = ra[i];
#pragma unroll
        for (int n = 0; n < 8; ++n) {             // pack 4 k-values -> b64 store
            uint2 pk;
            pk.x = get16(rb[0], n) | (get16(rb[1], n) << 16);
            pk.y = get16(rb[2], n) | (get16(rb[3], n) << 16);
            *(uint2*)(&Bs[buf][(bnq + n) * LD + bkq]) = pk;
        }
    };
    auto compute = [&](int buf) {
#pragma unroll
        for (int kc = 0; kc < 2; ++kc) {
            FragBF fa[2], fb[4];
            const int kb  = kc * 32 + (lane >> 4) * 8;   // A layout
            const int kbb = kc * 32 + (lane >> 4) * 8;   // B layout
#pragma unroll
            for (int mt = 0; mt < 2; ++mt) {
                const unsigned short* p = &As[buf][(wm * 32 + mt * 16 + (lane & 15)) * LD];
                fa[mt].q[0] = *(const uint4*)(p + kb);
                fa[mt].q[1] = *(const uint4*)(p + kb + 16);
            }
#pragma unroll
            for (int nt = 0; nt < 4; ++nt) {
                const unsigned short* p = &Bs[buf][(wn * 64 + nt * 16 + (lane & 15)) * LD];
                fb[nt].q[0] = *(const uint4*)(p + kbb);
                fb[nt].q[1] = *(const uint4*)(p + kbb + 16);
            }
#pragma unroll
            for (int mt = 0; mt < 2; ++mt)
#pragma unroll
                for (int nt = 0; nt < 4; ++nt)
                    acc[mt][nt] = wmma_bf16(fa[mt].v, fb[nt].v, acc[mt][nt]);
        }
    };

    // pipeline: prologue
    loadT(0);
    storeT(0);
    __syncthreads();
    int buf = 0;
    for (int k0 = 64; k0 < K; k0 += 64) {
        loadT(k0);                                 // global loads overlap compute
        if (k0 + 128 < K) {
            __builtin_prefetch(A + (size_t)(m0 + ar) * K + k0 + 64 + ak, 0, 3);
            __builtin_prefetch(B + (size_t)(k0 + 64 + bkq) * N + n0 + bnq, 0, 3);
        }
        compute(buf);
        storeT(buf ^ 1);
        __syncthreads();
        buf ^= 1;
    }
    compute(buf);

    // --- epilogue (C layout: VGPR v -> row (lane>>4)*8+v, col lane&15)
#pragma unroll
    for (int mt = 0; mt < 2; ++mt) {
#pragma unroll
        for (int nt = 0; nt < 4; ++nt) {
            const int rbase = m0 + wm * 32 + mt * 16 + (lane >> 4) * 8;
            const int col   = n0 + wn * 64 + nt * 16 + (lane & 15);
#pragma unroll
            for (int v = 0; v < 8; ++v) {
                const size_t idx = (size_t)(rbase + v) * N + col;
                const float a = acc[mt][nt][v];
                if constexpr (MODE == 0) {
                    outBF[idx] = f2bf(a);
                } else if constexpr (MODE == 1) {
                    outF[idx] = a + resid[idx];
                } else {  // MODE 2: silu(acc) * aux
                    const float u = bf2f(aux[idx]);
                    const float s = a / (1.f + __expf(-a));
                    outBF[idx] = f2bf(s * u);
                }
            }
        }
    }
}

// ---------------------------------------------------------------------------
// Flash attention (causal). Grid: (S/128, NH). Block: 256 (8 waves).
// Each wave owns 16 query rows; 64-key blocks double-buffered through LDS.
//  Q,K,V,O are bf16 in (S, NH*HD) layout.
// ---------------------------------------------------------------------------
__global__ __launch_bounds__(256)
void flash_attn(const unsigned short* __restrict__ Qg,
                const unsigned short* __restrict__ Kg,
                const unsigned short* __restrict__ Vg,
                unsigned short* __restrict__ Og) {
    constexpr int BM = 128, BN = 64;
    constexpr int LK = 136;  // Ks row stride (u16): 128 + 8 pad
    constexpr int LV = 72;   // Vs/Ps row stride (u16): 64 + 8 pad
    __shared__ unsigned short Ks[2][BN * LK];     // [t][d]   (B-frag for Q@K^T)
    __shared__ unsigned short Vs[2][HD2 * LV];    // [d][t]   (B-frag for P@V)
    __shared__ unsigned short Ps[8 * 16 * LV];    // per-wave P exchange [row][t]

    const int tid  = threadIdx.x;
    const int lane = tid & 31;
    const int wave = tid >> 5;
    const int m0   = blockIdx.x * BM;
    const int head = blockIdx.y;
    const int qrow0 = m0 + wave * 16;
    const size_t hoff = (size_t)head * HD2;

    // Q fragments for this wave's 16 rows: 4 chunks of K=32 over HD=128
    FragBF qf[4];
    {
        const int r  = qrow0 + (lane & 15);
        const int kb = (lane >> 4) * 8;
        const unsigned short* qp = Qg + (size_t)r * H_ + hoff;
#pragma unroll
        for (int kc = 0; kc < 4; ++kc) {
            qf[kc].q[0] = *(const uint4*)(qp + kc * 32 + kb);
            qf[kc].q[1] = *(const uint4*)(qp + kc * 32 + kb + 16);
        }
    }

    v8f o[8];
#pragma unroll
    for (int dt = 0; dt < 8; ++dt) o[dt] = v8f_zero();
    float mrow[8], lrow[8];
#pragma unroll
    for (int v = 0; v < 8; ++v) { mrow[v] = -1e30f; lrow[v] = 0.f; }

    const float scale = 0.08838834764831845f;  // 1/sqrt(128)
    const int nblocks = m0 / BN + 2;           // causal: keys up to m0+127

    // loader indices (lanes coalesced along d)
    const int rk = tid >> 2;                   // K row 0..63
    const int dk = (tid & 3) * 32;             // K d-chunk (u16)
    const int vtq = (tid >> 4) * 4;            // V: 4 consecutive t rows
    const int vdq = (tid & 15) * 8;            // V: 8 d values

    uint4 rkr[4], rvr[4];
    auto loadKV = [&](int t0) {
        const uint4* gk = (const uint4*)(Kg + (size_t)(t0 + rk) * H_ + hoff + dk);
#pragma unroll
        for (int i = 0; i < 4; ++i) rkr[i] = gk[i];
#pragma unroll
        for (int r = 0; r < 4; ++r)
            rvr[r] = *(const uint4*)(Vg + (size_t)(t0 + vtq + r) * H_ + hoff + vdq);
    };
    auto storeKV = [&](int buf) {
        unsigned short* kp = &Ks[buf][rk * LK + dk];
#pragma unroll
        for (int i = 0; i < 4; ++i) *(uint4*)(kp + 8 * i) = rkr[i];
#pragma unroll
        for (int n = 0; n < 8; ++n) {          // V transposed: pack 4 t -> b64
            uint2 pk;
            pk.x = get16(rvr[0], n) | (get16(rvr[1], n) << 16);
            pk.y = get16(rvr[2], n) | (get16(rvr[3], n) << 16);
            *(uint2*)(&Vs[buf][(vdq + n) * LV + vtq]) = pk;
        }
    };

    loadKV(0);
    storeKV(0);
    __syncthreads();
    int buf = 0;

    for (int j = 0; j < nblocks; ++j) {
        const int t0 = j * BN;
        const bool more = (j + 1) < nblocks;
        if (more) loadKV(t0 + BN);             // overlap with S compute

        // --- S = scale * Q @ K^T : kc-outer, 4 independent accumulator chains
        v8f s[4];
#pragma unroll
        for (int nt = 0; nt < 4; ++nt) s[nt] = v8f_zero();
#pragma unroll
        for (int kc = 0; kc < 4; ++kc) {
            FragBF kb_[4];
            const int kbb = kc * 32 + (lane >> 4) * 8;
#pragma unroll
            for (int nt = 0; nt < 4; ++nt) {
                const unsigned short* kp = &Ks[buf][(nt * 16 + (lane & 15)) * LK];
                kb_[nt].q[0] = *(const uint4*)(kp + kbb);
                kb_[nt].q[1] = *(const uint4*)(kp + kbb + 16);
            }
#pragma unroll
            for (int nt = 0; nt < 4; ++nt)
                s[nt] = wmma_bf16(qf[kc].v, kb_[nt].v, s[nt]);
        }

        // --- mask + online softmax (rows replicated across 16-lane half-wave)
        float p[4][8];
#pragma unroll
        for (int nt = 0; nt < 4; ++nt) {
            const int col = t0 + nt * 16 + (lane & 15);
#pragma unroll
            for (int v = 0; v < 8; ++v) {
                const int row = qrow0 + (lane >> 4) * 8 + v;
                p[nt][v] = (col <= row) ? s[nt][v] * scale : -1e30f;
            }
        }
#pragma unroll
        for (int v = 0; v < 8; ++v) {
            float mx = p[0][v];
#pragma unroll
            for (int nt = 1; nt < 4; ++nt) mx = fmaxf(mx, p[nt][v]);
#pragma unroll
            for (int msk = 1; msk < 16; msk <<= 1) mx = fmaxf(mx, __shfl_xor(mx, msk, 32));
            const float mnew = fmaxf(mrow[v], mx);
            const float alpha = __expf(mrow[v] - mnew);
            mrow[v] = mnew;
            float rs = 0.f;
#pragma unroll
            for (int nt = 0; nt < 4; ++nt) {
                p[nt][v] = __expf(p[nt][v] - mnew);
                rs += p[nt][v];
            }
#pragma unroll
            for (int msk = 1; msk < 16; msk <<= 1) rs += __shfl_xor(rs, msk, 32);
            lrow[v] = lrow[v] * alpha + rs;
#pragma unroll
            for (int dt = 0; dt < 8; ++dt) o[dt][v] *= alpha;
        }

        // park next K/V tile in LDS now -> frees staging regs before PV spike
        if (more) storeKV(buf ^ 1);

        // --- redistribute P (C layout -> A-fragment layout) via wave-local LDS
        unsigned short* pw = Ps + wave * 16 * LV;
#pragma unroll
        for (int nt = 0; nt < 4; ++nt)
#pragma unroll
            for (int v = 0; v < 8; ++v)
                pw[((lane >> 4) * 8 + v) * LV + nt * 16 + (lane & 15)] = f2bf(p[nt][v]);

        FragBF pf[2];
        {
            const unsigned short* pr = pw + (lane & 15) * LV;
            const int kb = (lane >> 4) * 8;
#pragma unroll
            for (int kc = 0; kc < 2; ++kc) {
                pf[kc].q[0] = *(const uint4*)(pr + kc * 32 + kb);
                pf[kc].q[1] = *(const uint4*)(pr + kc * 32 + kb + 16);
            }
        }

        // --- O += P @ V : kc-outer, 8 preloaded V-frags, 8 independent chains
#pragma unroll
        for (int kc = 0; kc < 2; ++kc) {
            FragBF vb[8];
            const int kbb = kc * 32 + (lane >> 4) * 8;
#pragma unroll
            for (int dt = 0; dt < 8; ++dt) {
                const unsigned short* vp = &Vs[buf][(dt * 16 + (lane & 15)) * LV];
                vb[dt].q[0] = *(const uint4*)(vp + kbb);
                vb[dt].q[1] = *(const uint4*)(vp + kbb + 16);
            }
#pragma unroll
            for (int dt = 0; dt < 8; ++dt)
                o[dt] = wmma_bf16(pf[kc].v, vb[dt].v, o[dt]);
        }
        __syncthreads();
        buf ^= 1;
    }

    // --- finalize: O /= l, store bf16
#pragma unroll
    for (int dt = 0; dt < 8; ++dt) {
#pragma unroll
        for (int v = 0; v < 8; ++v) {
            const int row = qrow0 + (lane >> 4) * 8 + v;
            const int col = head * HD2 + dt * 16 + (lane & 15);
            Og[(size_t)row * H_ + col] = f2bf(o[dt][v] / lrow[v]);
        }
    }
}

// ---------------------------------------------------------------------------
// Host-side orchestration
// ---------------------------------------------------------------------------
extern "C" void kernel_launch(void* const* d_in, const int* in_sizes, int n_in,
                              void* d_out, int out_size, void* d_ws, size_t ws_size,
                              hipStream_t stream) {
    (void)in_sizes; (void)n_in; (void)out_size; (void)ws_size;
    const float* x  = (const float*)d_in[0];
    const float* wq = (const float*)d_in[1];
    const float* wk = (const float*)d_in[2];
    const float* wv = (const float*)d_in[3];
    const float* wo = (const float*)d_in[4];
    const float* wg = (const float*)d_in[5];
    const float* wu = (const float*)d_in[6];
    const float* wd = (const float*)d_in[7];
    const float* g1 = (const float*)d_in[8];
    const float* g2 = (const float*)d_in[9];
    float* out = (float*)d_out;

    char* ws = (char*)d_ws;
    size_t off = 0;
    auto take = [&](size_t bytes) -> char* {
        char* p = ws + off;
        off += (bytes + 255) & ~(size_t)255;
        return p;
    };
    const size_t HH = (size_t)H_ * H_;
    const size_t HI = (size_t)H_ * I_;
    const size_t SH = (size_t)S_ * H_;
    const size_t SI = (size_t)S_ * I_;

    unsigned short* wq_bf = (unsigned short*)take(HH * 2);
    unsigned short* wk_bf = (unsigned short*)take(HH * 2);
    unsigned short* wv_bf = (unsigned short*)take(HH * 2);
    unsigned short* wo_bf = (unsigned short*)take(HH * 2);
    unsigned short* wg_bf = (unsigned short*)take(HI * 2);
    unsigned short* wu_bf = (unsigned short*)take(HI * 2);
    unsigned short* wd_bf = (unsigned short*)take(HI * 2);
    unsigned short* h_bf  = (unsigned short*)take(SH * 2);  // later reused as attn_bf
    unsigned short* q_bf  = (unsigned short*)take(SH * 2);  // later reused as h2_bf
    unsigned short* k_bf  = (unsigned short*)take(SH * 2);
    unsigned short* v_bf  = (unsigned short*)take(SH * 2);
    unsigned short* up_bf = (unsigned short*)take(SI * 2);
    unsigned short* m_bf  = (unsigned short*)take(SI * 2);
    float* x1 = (float*)take(SH * 4);

    // 1) weights -> bf16
    const int nHH4 = (int)(HH / 4), nHI4 = (int)(HI / 4);
    cvt_f32_to_bf16<<<nHH4 / 256, 256, 0, stream>>>(wq, wq_bf, nHH4);
    cvt_f32_to_bf16<<<nHH4 / 256, 256, 0, stream>>>(wk, wk_bf, nHH4);
    cvt_f32_to_bf16<<<nHH4 / 256, 256, 0, stream>>>(wv, wv_bf, nHH4);
    cvt_f32_to_bf16<<<nHH4 / 256, 256, 0, stream>>>(wo, wo_bf, nHH4);
    cvt_f32_to_bf16<<<nHI4 / 256, 256, 0, stream>>>(wg, wg_bf, nHI4);
    cvt_f32_to_bf16<<<nHI4 / 256, 256, 0, stream>>>(wu, wu_bf, nHI4);
    cvt_f32_to_bf16<<<nHI4 / 256, 256, 0, stream>>>(wd, wd_bf, nHI4);

    // 2) h = rmsnorm(x, g1)
    rmsnorm_to_bf16<<<S_, 256, 0, stream>>>(x, g1, h_bf);

    // 3) q/k/v = h @ {wq,wk,wv}
    dim3 gH(H_ / 128, S_ / 128);
    gemm_bf16<0><<<gH, 256, 0, stream>>>(h_bf, wq_bf, S_, H_, H_, q_bf, nullptr, nullptr, nullptr);
    gemm_bf16<0><<<gH, 256, 0, stream>>>(h_bf, wk_bf, S_, H_, H_, k_bf, nullptr, nullptr, nullptr);
    gemm_bf16<0><<<gH, 256, 0, stream>>>(h_bf, wv_bf, S_, H_, H_, v_bf, nullptr, nullptr, nullptr);

    // 4) RoPE on q, k (in-place)
    rope_inplace<<<(S_ * NH_ * 64) / 256, 256, 0, stream>>>(q_bf, k_bf);

    // 5) causal flash attention -> attn (reuse h_bf)
    unsigned short* attn_bf = h_bf;
    flash_attn<<<dim3(S_ / 128, NH_), 256, 0, stream>>>(q_bf, k_bf, v_bf, attn_bf);

    // 6) x1 = x + attn @ wo
    gemm_bf16<1><<<gH, 256, 0, stream>>>(attn_bf, wo_bf, S_, H_, H_, nullptr, x1, x, nullptr);

    // 7) h2 = rmsnorm(x1, g2)  (reuse q_bf)
    unsigned short* h2_bf = q_bf;
    rmsnorm_to_bf16<<<S_, 256, 0, stream>>>(x1, g2, h2_bf);

    // 8) up = h2 @ w_up ; m = silu(h2 @ w_gate) * up
    dim3 gI(I_ / 128, S_ / 128);
    gemm_bf16<0><<<gI, 256, 0, stream>>>(h2_bf, wu_bf, S_, I_, H_, up_bf, nullptr, nullptr, nullptr);
    gemm_bf16<2><<<gI, 256, 0, stream>>>(h2_bf, wg_bf, S_, I_, H_, m_bf, nullptr, nullptr, up_bf);

    // 9) out = x1 + m @ w_down
    gemm_bf16<1><<<gH, 256, 0, stream>>>(m_bf, wd_bf, S_, H_, I_, nullptr, out, x1, nullptr);
}
